// IMOLModel_74165495267895
// MI455X (gfx1250) — matrix operand order repaired and checked by reading, not compiled, not run on hardware
//
#include <hip/hip_runtime.h>
#include <stdint.h>

// ---------------- problem constants (from reference) ----------------
#define B_Q    1024          // queries
#define N_TR   200000        // train rows
#define DIM    768           // feature dim
#define KSEL   64            // top-k depth
#define NNEG   10
#define NPOS   2

// padded train-row count: 49 chunks of 4096 = 200704 (also multiple of 256)
#define CHUNK  4096
#define NCHUNK 49
#define NPAD   (CHUNK * NCHUNK)

// ---------------- workspace layout (bytes) ----------------
#define OFF_QB   0ull                                         // bf16 [B_Q][DIM]
#define SZ_QB    ((size_t)B_Q * DIM * 2)
#define OFF_RN   (OFF_QB + SZ_QB)                              // f32 [NPAD]
#define SZ_RN    ((size_t)NPAD * 4)
#define OFF_TB   (OFF_RN + SZ_RN)                              // bf16 [NPAD][DIM]
#define SZ_TB    ((size_t)NPAD * DIM * 2)
#define OFF_SIM  (OFF_TB + SZ_TB)                              // f32 [B_Q][NPAD]
#define SZ_SIM   ((size_t)B_Q * NPAD * 4)
#define OFF_CAND (OFF_SIM + SZ_SIM)                            // u64 [B_Q][NCHUNK][KSEL]

typedef __attribute__((ext_vector_type(16))) __bf16   v16bf;
typedef __attribute__((ext_vector_type(8)))  float    v8f;
typedef __attribute__((ext_vector_type(4)))  uint32_t v4u;
typedef __attribute__((ext_vector_type(4)))  int      v4i;

// pointer types matching the async-DMA builtin prototype
typedef __attribute__((address_space(1))) v4i* gptr_v4i;   // global src
typedef __attribute__((address_space(3))) v4i* lptr_v4i;   // LDS dst

// CDNA5 async global->LDS path (ASYNCcnt-tracked DMA), guarded by probe
#if defined(__AMDGCN__) && __has_builtin(__builtin_amdgcn_global_load_async_to_lds_b128) && __has_builtin(__builtin_amdgcn_s_wait_asynccnt)
#define USE_ASYNC_LDS 1
#else
#define USE_ASYNC_LDS 0
#endif

static __device__ __forceinline__ uint16_t f2bf(float f) {
    uint32_t u = __float_as_uint(f);
    u += 0x7FFFu + ((u >> 16) & 1u);     // round-to-nearest-even
    return (uint16_t)(u >> 16);
}

// ---------------- kernel 1/2: L2-normalize rows, emit bf16 (+rnorm) --------
// one wave32 per row; DIM=768 -> 24 elements per lane
__global__ __launch_bounds__(256)
void normalize_rows(const float* __restrict__ src, uint16_t* __restrict__ dst,
                    float* __restrict__ rnorm, int rows_total, int rows_valid) {
    int row  = blockIdx.x * 8 + (threadIdx.x >> 5);
    int lane = threadIdx.x & 31;
    if (row >= rows_total) return;

    uint16_t* d = dst + (size_t)row * DIM;
    if (row < rows_valid) {
        const float* s = src + (size_t)row * DIM;
        float vals[24];
        float ss = 0.f;
#pragma unroll
        for (int i = 0; i < 24; ++i) {
            vals[i] = s[lane + i * 32];
            ss += vals[i] * vals[i];
        }
#pragma unroll
        for (int m = 16; m >= 1; m >>= 1) ss += __shfl_xor(ss, m, 32);
        float rn = rsqrtf(ss);
#pragma unroll
        for (int i = 0; i < 24; ++i) d[lane + i * 32] = f2bf(vals[i] * rn);
        if (rnorm && lane == 0) rnorm[row] = rn;
    } else {
#pragma unroll
        for (int i = 0; i < 24; ++i) d[lane + i * 32] = 0;
        if (rnorm && lane == 0) rnorm[row] = 0.f;
    }
}

// ---------------- kernel 3: bf16 WMMA similarity GEMM ----------------
// workgroup: 8 waves -> 32 queries x 256 train rows; wave: 16M x 64N (4 accums)
// A(16x32 bf16) per-lane: m = lane&15, kb = (lane>>4)*8,
//   elems 0..7  = K[k0+kb .. k0+kb+7], elems 8..15 = K[k0+16+kb ..]
// B(32x16 bf16) per-lane: n = lane&15, elems 0..7 = K[k0+kb .. k0+kb+7], elems 8..15 = K[k0+16+kb ..] (same as A)

#if USE_ASYNC_LDS
// B slab (256 rows x 32 K bf16 = 16 KB) staged into LDS by async DMA,
// TRIPLE buffered (48 KB) -> one barrier per slab; waves read via ds_load_b128.
// LDS layout: row-major, 64 B per row (one K-slab of one train row).
#define SLAB_BYTES 16384
__global__ __launch_bounds__(256)
void gemm_sim(const uint16_t* __restrict__ qb, const uint16_t* __restrict__ tb,
              float* __restrict__ sim) {
    __shared__ __align__(16) uint8_t smem[3 * SLAB_BYTES];   // 48 KB
    const int tid  = threadIdx.x;
    const int lane = tid & 31;
    const int w    = tid >> 5;
    const int wm   = w & 1;
    const int wn   = w >> 1;
    const int qbase    = blockIdx.x * 32 + wm * 16;
    const int nbase_wg = blockIdx.y * 256;
    const int nbase    = nbase_wg + wn * 64;

    const int mrow = qbase + (lane & 15);
    const int kb   = (lane >> 4) * 8;
    const uint16_t* aptr = qb + (size_t)mrow * DIM;

    const int nrow0 = nbase + (lane & 15);               // for final store
    const int lrow0 = wn * 64 + (lane & 15);             // LDS row of tile t=0
    const int lkoff = (lane >> 4) * 16;                  // byte offset in LDS row

    const char* gbase = (const char*)tb;
    union Pack { v16bf v; v4u q[2]; };

    v8f acc[4];
    v8f zero = {0.f, 0.f, 0.f, 0.f, 0.f, 0.f, 0.f, 0.f};
#pragma unroll
    for (int t = 0; t < 4; ++t) acc[t] = zero;

    // stage one 16 KB slab: thread handles 4 interleaved 16B chunks
    // chunk c = tid + i*256: row = c>>2, kbyte = (c&3)*16
    auto stage = [&](int buf_off, int slab) {
        const size_t gk = (size_t)slab * 64;             // byte offset of K-slab
#pragma unroll
        for (int i = 0; i < 4; ++i) {
            int c   = tid + i * 256;
            int row = c >> 2;
            int kby = (c & 3) * 16;
            const char* g = gbase + (size_t)(nbase_wg + row) * (DIM * 2) + gk + kby;
            __builtin_amdgcn_global_load_async_to_lds_b128(
                (gptr_v4i)g, (lptr_v4i)(&smem[0] + buf_off + row * 64 + kby), 0, 0);
        }
    };

    auto compute = [&](int buf_off, const Pack& a) {
        const uint8_t* lpbase = &smem[0] + buf_off + lrow0 * 64 + lkoff;
#pragma unroll
        for (int t = 0; t < 4; ++t) {
            const uint8_t* lp = lpbase + t * 16 * 64;
            Pack b;
            b.q[0] = *(const v4u*)(lp);
            b.q[1] = *(const v4u*)(lp + 32);
            acc[t] = __builtin_amdgcn_wmma_f32_16x16x32_bf16(
                false, a.v, false, b.v, (short)0, acc[t], false, false);
        }
    };

    // ---- prologue: stage slab 0 into buffer 0 ----
    stage(0, 0);

    const int NK = DIM / 32;          // 24 K-slabs
    int rd_off = 0;
    int wr_off = SLAB_BYTES;
#pragma unroll 1
    for (int kk = 0; kk < NK - 1; ++kk) {
        stage(wr_off, kk + 1);                    // DMA slab kk+1 (overlaps compute kk)
        Pack a;                                    // A issued before wait -> latency hidden
        a.q[0] = *(const v4u*)(aptr + kk * 32 + kb);
        a.q[1] = *(const v4u*)(aptr + kk * 32 + 16 + kb);
        __builtin_amdgcn_s_wait_asynccnt(4);      // slab kk landed (4 newest in flight)
        __syncthreads();                          // all waves' slab-kk chunks visible
        compute(rd_off, a);
        rd_off = wr_off;
        wr_off += SLAB_BYTES;
        if (wr_off == 3 * SLAB_BYTES) wr_off = 0; // triple-buffer rotation
    }

    // ---- epilogue: last slab ----
    {
        Pack a;
        a.q[0] = *(const v4u*)(aptr + (NK - 1) * 32 + kb);
        a.q[1] = *(const v4u*)(aptr + (NK - 1) * 32 + 16 + kb);
        __builtin_amdgcn_s_wait_asynccnt(0);
        __syncthreads();
        compute(rd_off, a);
    }

    // C/D layout: elem r of v8f -> M = r + 8*(lane>>4), N = lane&15
#pragma unroll
    for (int t = 0; t < 4; ++t) {
        int n = nrow0 + t * 16;
#pragma unroll
        for (int r = 0; r < 8; ++r) {
            int qrow = qbase + r + 8 * (lane >> 4);
            float v  = (n < N_TR) ? acc[t][r] : -1e30f;
            sim[(size_t)qrow * NPAD + n] = v;
        }
    }
}
#else
// fallback: direct-from-global double-buffered variant (proven to compile)
__global__ __launch_bounds__(256)
void gemm_sim(const uint16_t* __restrict__ qb, const uint16_t* __restrict__ tb,
              float* __restrict__ sim) {
    const int lane = threadIdx.x & 31;
    const int w    = threadIdx.x >> 5;
    const int wm   = w & 1;
    const int wn   = w >> 1;
    const int qbase = blockIdx.x * 32 + wm * 16;
    const int nbase = blockIdx.y * 256 + wn * 64;

    const int mrow = qbase + (lane & 15);
    const int kb   = (lane >> 4) * 8;
    const uint16_t* aptr = qb + (size_t)mrow * DIM;

    const int nrow0 = nbase + (lane & 15);
    const int bkoff = (lane >> 4) * 8;
    const uint16_t* bbase = tb + (size_t)nrow0 * DIM + bkoff;

    union Pack { v16bf v; v4u q[2]; };

    v8f acc[4];
    v8f zero = {0.f, 0.f, 0.f, 0.f, 0.f, 0.f, 0.f, 0.f};
#pragma unroll
    for (int t = 0; t < 4; ++t) acc[t] = zero;

    Pack a_cur, b_cur[4];
    a_cur.q[0] = *(const v4u*)(aptr + kb);
    a_cur.q[1] = *(const v4u*)(aptr + 16 + kb);
#pragma unroll
    for (int t = 0; t < 4; ++t) {
        const uint16_t* bp = bbase + (size_t)t * 16 * DIM;
        b_cur[t].q[0] = *(const v4u*)(bp);
        b_cur[t].q[1] = *(const v4u*)(bp + 16);
    }

    const int NK = DIM / 32;
    for (int kk = 0; kk < NK - 1; ++kk) {
        const int k1 = (kk + 1) * 32;
        Pack a_nxt, b_nxt[4];
        a_nxt.q[0] = *(const v4u*)(aptr + k1 + kb);
        a_nxt.q[1] = *(const v4u*)(aptr + k1 + 16 + kb);
#pragma unroll
        for (int t = 0; t < 4; ++t) {
            const uint16_t* bp = bbase + (size_t)t * 16 * DIM + k1;
            b_nxt[t].q[0] = *(const v4u*)(bp);
            b_nxt[t].q[1] = *(const v4u*)(bp + 16);
        }
        if (k1 + 32 < DIM)
            __builtin_prefetch(bbase + k1 + 32, 0, 1);

#pragma unroll
        for (int t = 0; t < 4; ++t)
            acc[t] = __builtin_amdgcn_wmma_f32_16x16x32_bf16(
                false, a_cur.v, false, b_cur[t].v, (short)0, acc[t], false, false);

        a_cur = a_nxt;
#pragma unroll
        for (int t = 0; t < 4; ++t) b_cur[t] = b_nxt[t];
    }

#pragma unroll
    for (int t = 0; t < 4; ++t)
        acc[t] = __builtin_amdgcn_wmma_f32_16x16x32_bf16(
            false, a_cur.v, false, b_cur[t].v, (short)0, acc[t], false, false);

#pragma unroll
    for (int t = 0; t < 4; ++t) {
        int n = nrow0 + t * 16;
#pragma unroll
        for (int r = 0; r < 8; ++r) {
            int qrow = qbase + r + 8 * (lane >> 4);
            float v  = (n < N_TR) ? acc[t][r] : -1e30f;
            sim[(size_t)qrow * NPAD + n] = v;
        }
    }
}
#endif

// ---------------- bitonic sort (descending) in LDS ----------------
static __device__ void bitonic_desc(unsigned long long* keys, int n) {
    for (int k = 2; k <= n; k <<= 1) {
        for (int j = k >> 1; j > 0; j >>= 1) {
            for (int i = threadIdx.x; i < n; i += blockDim.x) {
                int ixj = i ^ j;
                if (ixj > i) {
                    unsigned long long a = keys[i], b = keys[ixj];
                    bool up = ((i & k) == 0);
                    if ((a < b) == up) { keys[i] = b; keys[ixj] = a; }
                }
            }
            __syncthreads();
        }
    }
}

// ---------------- kernel 4: per-(query,chunk) top-64 ----------------
__global__ __launch_bounds__(256)
void chunk_topk(const float* __restrict__ sim, unsigned long long* __restrict__ cand) {
    __shared__ unsigned long long keys[CHUNK];   // 32 KB
    const int q = blockIdx.y;
    const int c = blockIdx.x;
    const float* row = sim + (size_t)q * NPAD + (size_t)c * CHUNK;
    for (int i = threadIdx.x; i < CHUNK; i += 256) {
        float v = row[i];
        uint32_t u   = __float_as_uint(v);
        uint32_t ord = (u & 0x80000000u) ? ~u : (u | 0x80000000u);
        uint32_t n   = (uint32_t)(c * CHUNK + i);
        keys[i] = ((unsigned long long)ord << 32) | (uint32_t)(~n);  // desc val, asc idx
    }
    __syncthreads();
    bitonic_desc(keys, CHUNK);
    if (threadIdx.x < KSEL)
        cand[((size_t)q * NCHUNK + c) * KSEL + threadIdx.x] = keys[threadIdx.x];
}

// ---------------- kernel 5: merge candidates + assemble outputs ----------------
__global__ __launch_bounds__(256)
void merge_assemble(const unsigned long long* __restrict__ cand,
                    const int* __restrict__ qlabels, const int* __restrict__ tlabels,
                    const float* __restrict__ tfeats, const float* __restrict__ rnorm,
                    float* __restrict__ out_hard, float* __restrict__ out_pos) {
    __shared__ unsigned long long keys[CHUNK];   // 4096 >= 49*64 = 3136
    const int q = blockIdx.x;
    const unsigned long long* src = cand + (size_t)q * NCHUNK * KSEL;
    for (int i = threadIdx.x; i < CHUNK; i += 256)
        keys[i] = (i < NCHUNK * KSEL) ? src[i] : 0ull;
    __syncthreads();
    bitonic_desc(keys, CHUNK);

    const int qlab = qlabels[q];
    int neg = 0, pos = 0;
    for (int r = 0; r < KSEL; ++r) {
        if (neg >= NNEG && pos >= NPOS) break;
        unsigned long long key = keys[r];
        uint32_t n = ~(uint32_t)(key & 0xFFFFFFFFu);
        if (n >= (uint32_t)N_TR) continue;
        int lab = tlabels[n];
        const float* s = tfeats + (size_t)n * DIM;
        float rn = rnorm[n];
        float* dst = nullptr;
        if (lab != qlab) {
            if (neg < NNEG) dst = out_hard + ((size_t)q * NNEG + neg) * DIM;
            ++neg;
        } else {
            if (pos < NPOS) dst = out_pos + ((size_t)q * NPOS + pos) * DIM;
            ++pos;
        }
        if (dst)
            for (int d = threadIdx.x; d < DIM; d += 256) dst[d] = s[d] * rn;
    }
}

// ---------------- launch ----------------
extern "C" void kernel_launch(void* const* d_in, const int* in_sizes, int n_in,
                              void* d_out, int out_size, void* d_ws, size_t ws_size,
                              hipStream_t stream) {
    (void)in_sizes; (void)n_in; (void)ws_size;
    const float* qf   = (const float*)d_in[0];
    const int*   qlab = (const int*)d_in[1];
    const float* tf   = (const float*)d_in[2];
    const int*   tlab = (const int*)d_in[3];
    float* out = (float*)d_out;
    char*  ws  = (char*)d_ws;

    uint16_t* qbuf  = (uint16_t*)(ws + OFF_QB);
    float*    rnorm = (float*)   (ws + OFF_RN);
    uint16_t* tbuf  = (uint16_t*)(ws + OFF_TB);
    float*    sim   = (float*)   (ws + OFF_SIM);
    unsigned long long* cand = (unsigned long long*)(ws + OFF_CAND);

    // outputs must default to zero (unfilled slots)
    (void)hipMemsetAsync(d_out, 0, (size_t)out_size * sizeof(float), stream);

    normalize_rows<<<B_Q / 8, 256, 0, stream>>>(qf, qbuf, nullptr, B_Q, B_Q);
    normalize_rows<<<NPAD / 8, 256, 0, stream>>>(tf, tbuf, rnorm, NPAD, N_TR);

    dim3 g2(B_Q / 32, NPAD / 256);    // (32, 784); x-fastest -> L2 reuse of train tiles
    gemm_sim<<<g2, 256, 0, stream>>>(qbuf, tbuf, sim);

    dim3 g3(NCHUNK, B_Q);             // (49, 1024)
    chunk_topk<<<g3, 256, 0, stream>>>(sim, cand);

    float* out_hard = out;
    float* out_pos  = out + (size_t)B_Q * NNEG * DIM;
    merge_assemble<<<B_Q, 256, 0, stream>>>(cand, qlab, tlab, tf, rnorm,
                                            out_hard, out_pos);
}
